// MultiheadAttention_58669253263910
// MI455X (gfx1250) — matrix operand order repaired and checked by reading, compiled, not run on hardware
//
#include <hip/hip_runtime.h>
#include <hip/hip_bf16.h>
#include <cstdint>
#include <cstddef>

// ---------------------------------------------------------------------------
// MultiheadAttention forward for MI455X (gfx1250).
// bf16 WMMA + double-buffered async global->LDS staging (ASYNCcnt), k=64 stages,
// explicit 2x pipeline unroll; all fragment loads hoisted ahead of WMMA bursts.
// B=2, S=2048, D=1024, H=16, hd=64.  Workspace ~50.4 MB.
// ---------------------------------------------------------------------------

typedef __bf16 bf16;
typedef __attribute__((ext_vector_type(16))) __bf16 v16bf;
typedef __attribute__((ext_vector_type(8)))  __bf16 v8bf;
typedef __attribute__((ext_vector_type(8)))  float  v8f;

#define BDIM 2
#define SDIM 2048
#define DDIM 1024
#define HDIM 16
#define HD   64
#define LDP  72   // padded LDS row length (144B): conflict-free, 16B-aligned

static __device__ __forceinline__ v16bf make16(v8bf lo, v8bf hi) {
  v16bf r;
#pragma unroll
  for (int i = 0; i < 8; ++i) { r[i] = lo[i]; r[i + 8] = hi[i]; }
  return r;
}

// A-fragment (16x32 bf16, row-major, ld ldk): lane m=l&15, half=l>>4;
// elems 0..7 <- k[half*8..+8), elems 8..15 <- k[16+half*8..+8).
static __device__ __forceinline__ v16bf load_a_frag(const bf16* base, int ldk) {
  int lane = threadIdx.x & 31;
  int m = lane & 15, half = lane >> 4;
  const bf16* p = base + (size_t)m * ldk + half * 8;
  v8bf c0 = *(const v8bf*)p;
  v8bf c1 = *(const v8bf*)(p + 16);
  return make16(c0, c1);
}

// B-fragment (32x16 bf16) for D=A*B with B=W^T, W row-major [n][k], ld ldk:
// lane n=l&15, half=l>>4; elems 0..7 <- k[half*8..+8), elems 8..15 <- k[16+half*8..+8).
static __device__ __forceinline__ v16bf load_b_frag(const bf16* base, int ldk) {
  int lane = threadIdx.x & 31;
  int n = lane & 15, half = lane >> 4;
  const bf16* p = base + (size_t)n * ldk + half * 8;
  return make16(*(const v8bf*)p, *(const v8bf*)(p + 16));
}

static __device__ __forceinline__ v8f wmma_bf16(v16bf a, v16bf b, v8f c) {
  return __builtin_amdgcn_wmma_f32_16x16x32_bf16(false, a, false, b, (short)0, c, false, false);
}

// Async global -> LDS, 16 bytes, tracked with ASYNCcnt.
static __device__ __forceinline__ void async_b128(void* lds, const void* g) {
  asm volatile("global_load_async_to_lds_b128 %0, %1, off"
               :: "v"((uint32_t)(uintptr_t)lds), "v"(g) : "memory");
}
static __device__ __forceinline__ void wait_async() {
  asm volatile("s_wait_asynccnt 0" ::: "memory");
}

// ---------------------------------------------------------------------------
__global__ __launch_bounds__(256) void cvt_f32_to_bf16(const float* __restrict__ src,
                                                       bf16* __restrict__ dst, int n) {
  int i = (blockIdx.x * 256 + threadIdx.x) * 4;
  if (i + 3 < n) {
    float4 v = *(const float4*)(src + i);
    dst[i + 0] = (bf16)v.x;
    dst[i + 1] = (bf16)v.y;
    dst[i + 2] = (bf16)v.z;
    dst[i + 3] = (bf16)v.w;
  }
}

// ---------------------------------------------------------------------------
// Stage a 128x64 bf16 tile (rows padded to LDP). 1024 16B chunks, 256 threads
// -> 4 async loads per matrix per thread.
static __device__ __forceinline__ void stage_gemm64(bf16 (*at)[LDP], bf16 (*bt)[LDP],
                                                    const bf16* Ag, const bf16* Bg, int tid) {
#pragma unroll
  for (int i = 0; i < 4; ++i) {
    int c = tid * 4 + i;             // 0..1023
    int row = c >> 3, part = c & 7;  // 8 x 16B per 64-elem row
    async_b128(&at[row][part * 8], Ag + (size_t)row * DDIM + part * 8);
    async_b128(&bt[row][part * 8], Bg + (size_t)row * DDIM + part * 8);
  }
}

// One k=64 pipeline stage: drain DMA, barrier, kick next stage's DMA, 16 WMMAs.
// All 6 fragments of each k-sub are loaded before the 8-WMMA burst so ds_loads
// stay in flight (distinct dest regs -> partial s_wait_dscnt, not full drains).
static __device__ __forceinline__ void gemm_step(const bf16 (*At)[LDP], const bf16 (*Bt)[LDP],
                                                 bf16 (*Atn)[LDP], bf16 (*Btn)[LDP],
                                                 const bf16* AgN, const bf16* BgN, bool doStage,
                                                 int wm, int wn, int tid, v8f (&acc)[2][4]) {
  wait_async();
  __syncthreads();
  if (doStage) stage_gemm64(Atn, Btn, AgN, BgN, tid);
#pragma unroll
  for (int ksub = 0; ksub < 64; ksub += 32) {
    v16bf a0 = load_a_frag(&At[wm * 32][ksub], LDP);
    v16bf a1 = load_a_frag(&At[wm * 32 + 16][ksub], LDP);
    v16bf bfr[4];
#pragma unroll
    for (int nt = 0; nt < 4; ++nt)
      bfr[nt] = load_b_frag(&Bt[wn * 64 + nt * 16][ksub], LDP);
#pragma unroll
    for (int nt = 0; nt < 4; ++nt) {
      acc[0][nt] = wmma_bf16(a0, bfr[nt], acc[0][nt]);
      acc[1][nt] = wmma_bf16(a1, bfr[nt], acc[1][nt]);
    }
  }
}

// QKV projection: [4096,1024] @ Wqkv^T + bqkv -> head-major Q/K, transposed V.
__global__ __launch_bounds__(256) void qkv_gemm(const bf16* __restrict__ xb,
                                                const bf16* __restrict__ wb,
                                                const float* __restrict__ bqkv,
                                                bf16* __restrict__ Qh,
                                                bf16* __restrict__ Kh,
                                                bf16* __restrict__ Vth) {
  __shared__ __align__(16) bf16 Atile[2][128][LDP];
  __shared__ __align__(16) bf16 Btile[2][128][LDP];
  int tid = threadIdx.x;
  int lane = tid & 31, w = tid >> 5;
  int half = lane >> 4, ln = lane & 15;
  int wm = w & 3, wn = w >> 2;
  int m0 = blockIdx.x * 128 + wm * 32;
  int n0 = blockIdx.y * 128 + wn * 64;
  const bf16* Ag = xb + (size_t)(blockIdx.x * 128) * DDIM;
  const bf16* Bg = wb + (size_t)(blockIdx.y * 128) * DDIM;

  stage_gemm64(Atile[0], Btile[0], Ag, Bg, tid);
  v8f acc[2][4] = {};
  for (int s = 0; s < DDIM / 64; s += 2) {   // 16 stages, 2x unrolled
    gemm_step(Atile[0], Btile[0], Atile[1], Btile[1],
              Ag + (s + 1) * 64, Bg + (s + 1) * 64, true, wm, wn, tid, acc);
    gemm_step(Atile[1], Btile[1], Atile[0], Btile[0],
              Ag + (s + 2) * 64, Bg + (s + 2) * 64, s + 2 < DDIM / 64, wm, wn, tid, acc);
  }

  // Epilogue: bias + scatter. n -> h=n/192, r=n%192; r<64:Q, <128:K, else V^T.
#pragma unroll
  for (int mi = 0; mi < 2; ++mi) {
#pragma unroll
    for (int nt = 0; nt < 4; ++nt) {
      int n = n0 + nt * 16 + ln;
      float bias = bqkv[n];
      int h = n / 192, rr = n % 192;
#pragma unroll
      for (int r = 0; r < 8; ++r) {
        int m = m0 + mi * 16 + r + 8 * half;
        int b = m >> 11, sidx = m & (SDIM - 1);
        bf16 bv = (bf16)(acc[mi][nt][r] + bias);
        size_t bhs = (size_t)(b * HDIM + h) * SDIM + sidx;
        if (rr < 64)        Qh[bhs * HD + rr] = bv;
        else if (rr < 128)  Kh[bhs * HD + (rr - 64)] = bv;
        else                Vth[((size_t)(b * HDIM + h) * HD + (rr - 128)) * SDIM + sidx] = bv;
      }
    }
  }
}

// ---------------------------------------------------------------------------
// Stage 64x64 K tile and 64x64 V^T tile (rows padded to LDP).
static __device__ __forceinline__ void stage_attn64(bf16 (*kt)[LDP], bf16 (*vt)[LDP],
                                                    const bf16* Kp, const bf16* Vp,
                                                    int kb, int tid) {
#pragma unroll
  for (int i = 0; i < 4; ++i) {
    int c = tid * 4 + i;           // 0..511
    int r = c >> 3, p = c & 7;     // 8 x 16B per 64-elem row
    async_b128(&kt[r][p * 8], Kp + (size_t)(kb + r) * HD + p * 8);
    async_b128(&vt[r][p * 8], Vp + (size_t)r * SDIM + kb + p * 8);
  }
}

// One 64-key flash stage: 8 QK WMMAs, fused softmax over 64 keys, 8 PV WMMAs.
static __device__ __forceinline__ void attn_step(const bf16 (*Kt)[LDP], const bf16 (*Vt)[LDP],
                                                 bf16 (*Ktn)[LDP], bf16 (*Vtn)[LDP],
                                                 const bf16* Kp, const bf16* Vp,
                                                 int kbN, bool doStage,
                                                 v16bf qf0, v16bf qf1,
                                                 v8f (&acc)[4], float (&mrun)[8], float (&lrun)[8],
                                                 bf16 (*pw)[LDP], int tid, int half, int ln) {
  wait_async();
  __syncthreads();
  if (doStage) stage_attn64(Ktn, Vtn, Kp, Vp, kbN, tid);

  // logits: all 8 K-frags loaded first, then the 8-WMMA burst
  v16bf kf[4][2];
#pragma unroll
  for (int j = 0; j < 4; ++j) {
    kf[j][0] = load_b_frag(&Kt[j * 16][0], LDP);
    kf[j][1] = load_b_frag(&Kt[j * 16][32], LDP);
  }
  v8f sfr[4];
#pragma unroll
  for (int j = 0; j < 4; ++j) {
    v8f s = {};
    s = wmma_bf16(qf0, kf[j][0], s);
    s = wmma_bf16(qf1, kf[j][1], s);
    sfr[j] = s;
  }

  // online softmax across all 64 keys at once
  float alpha[8];
#pragma unroll
  for (int r = 0; r < 8; ++r) {
    float e0 = sfr[0][r] * 0.125f, e1 = sfr[1][r] * 0.125f;
    float e2 = sfr[2][r] * 0.125f, e3 = sfr[3][r] * 0.125f;
    float vmx = fmaxf(fmaxf(e0, e1), fmaxf(e2, e3));
#pragma unroll
    for (int off = 1; off < 16; off <<= 1) vmx = fmaxf(vmx, __shfl_xor(vmx, off, 32));
    float mnew = fmaxf(mrun[r], vmx);
    alpha[r] = __expf(mrun[r] - mnew);
    float p0 = __expf(e0 - mnew), p1 = __expf(e1 - mnew);
    float p2 = __expf(e2 - mnew), p3 = __expf(e3 - mnew);
    float rs = (p0 + p1) + (p2 + p3);
#pragma unroll
    for (int off = 1; off < 16; off <<= 1) rs += __shfl_xor(rs, off, 32);
    lrun[r] = lrun[r] * alpha[r] + rs;
    mrun[r] = mnew;
    sfr[0][r] = p0; sfr[1][r] = p1; sfr[2][r] = p2; sfr[3][r] = p3;
  }

  // P (16x64): C-frag layout -> wave-private LDS -> two A-frags.
  // Same-wave LDS ops are in-order; wave_barrier stops compiler reordering.
#pragma unroll
  for (int j = 0; j < 4; ++j)
#pragma unroll
    for (int r = 0; r < 8; ++r)
      pw[r + 8 * half][j * 16 + ln] = (bf16)sfr[j][r];
  __builtin_amdgcn_wave_barrier();
  v16bf pf0 = load_a_frag(&pw[0][0], LDP);
  v16bf pf1 = load_a_frag(&pw[0][32], LDP);
  __builtin_amdgcn_wave_barrier();

  // acc = alpha*acc + P @ V: V-frags loaded ahead of the rescale VALU so the
  // ds latency overlaps; then the 8-WMMA burst.
  v16bf vf[4][2];
#pragma unroll
  for (int nt = 0; nt < 4; ++nt) {
    vf[nt][0] = load_b_frag(&Vt[nt * 16][0], LDP);
    vf[nt][1] = load_b_frag(&Vt[nt * 16][32], LDP);
  }
#pragma unroll
  for (int nt = 0; nt < 4; ++nt)
#pragma unroll
    for (int r = 0; r < 8; ++r) acc[nt][r] *= alpha[r];
#pragma unroll
  for (int nt = 0; nt < 4; ++nt) {
    acc[nt] = wmma_bf16(pf0, vf[nt][0], acc[nt]);
    acc[nt] = wmma_bf16(pf1, vf[nt][1], acc[nt]);
  }
}

// Flash attention: block (4 waves) per (b,h,64-query tile); wave owns 16 rows.
__global__ __launch_bounds__(128) void attn_fwd(const bf16* __restrict__ Qh,
                                                const bf16* __restrict__ Kh,
                                                const bf16* __restrict__ Vth,
                                                bf16* __restrict__ vals) {
  __shared__ __align__(16) bf16 Ktile[2][64][LDP];
  __shared__ __align__(16) bf16 Vtile[2][64][LDP];
  __shared__ __align__(16) bf16 pshm[4][16][LDP];
  int tid = threadIdx.x;
  int lane = tid & 31, w = tid >> 5;
  int half = lane >> 4, ln = lane & 15;
  int bh = blockIdx.x >> 5;   // S/64 = 32 query tiles per (b,h)
  int qt = blockIdx.x & 31;
  const bf16* Qp = Qh + (size_t)bh * SDIM * HD;
  const bf16* Kp = Kh + (size_t)bh * SDIM * HD;
  const bf16* Vp = Vth + (size_t)bh * HD * SDIM;
  int qbase = qt * 64 + w * 16;

  v16bf qf0 = load_a_frag(Qp + (size_t)qbase * HD + 0, HD);
  v16bf qf1 = load_a_frag(Qp + (size_t)qbase * HD + 32, HD);

  stage_attn64(Ktile[0], Vtile[0], Kp, Vp, 0, tid);

  v8f acc[4] = {};
  float mrun[8], lrun[8];
#pragma unroll
  for (int r = 0; r < 8; ++r) { mrun[r] = -3.0e38f; lrun[r] = 0.f; }

  for (int s = 0; s < SDIM / 64; s += 2) {   // 32 stages, 2x unrolled
    attn_step(Ktile[0], Vtile[0], Ktile[1], Vtile[1], Kp, Vp,
              (s + 1) * 64, true, qf0, qf1, acc, mrun, lrun, pshm[w], tid, half, ln);
    attn_step(Ktile[1], Vtile[1], Ktile[0], Vtile[0], Kp, Vp,
              (s + 2) * 64, s + 2 < SDIM / 64, qf0, qf1, acc, mrun, lrun, pshm[w], tid, half, ln);
  }

  int b = bh >> 4, h = bh & 15;
#pragma unroll
  for (int nt = 0; nt < 4; ++nt) {
#pragma unroll
    for (int r = 0; r < 8; ++r) {
      int srow = qbase + r + 8 * half;
      float ov = acc[nt][r] / lrun[r];
      vals[((size_t)(b * SDIM + srow)) * DDIM + h * HD + nt * 16 + ln] = (bf16)ov;
    }
  }
}

// ---------------------------------------------------------------------------
// Output projection: vals[4096,1024] @ Wo^T + bo -> f32 out.
__global__ __launch_bounds__(256) void out_proj(const bf16* __restrict__ vb,
                                                const bf16* __restrict__ wob,
                                                const float* __restrict__ bo,
                                                float* __restrict__ out) {
  __shared__ __align__(16) bf16 Atile[2][128][LDP];
  __shared__ __align__(16) bf16 Btile[2][128][LDP];
  int tid = threadIdx.x;
  int lane = tid & 31, w = tid >> 5;
  int half = lane >> 4, ln = lane & 15;
  int wm = w & 3, wn = w >> 2;
  int m0 = blockIdx.x * 128 + wm * 32;
  int n0 = blockIdx.y * 128 + wn * 64;
  const bf16* Ag = vb + (size_t)(blockIdx.x * 128) * DDIM;
  const bf16* Bg = wob + (size_t)(blockIdx.y * 128) * DDIM;

  stage_gemm64(Atile[0], Btile[0], Ag, Bg, tid);
  v8f acc[2][4] = {};
  for (int s = 0; s < DDIM / 64; s += 2) {
    gemm_step(Atile[0], Btile[0], Atile[1], Btile[1],
              Ag + (s + 1) * 64, Bg + (s + 1) * 64, true, wm, wn, tid, acc);
    gemm_step(Atile[1], Btile[1], Atile[0], Btile[0],
              Ag + (s + 2) * 64, Bg + (s + 2) * 64, s + 2 < DDIM / 64, wm, wn, tid, acc);
  }
#pragma unroll
  for (int mi = 0; mi < 2; ++mi) {
#pragma unroll
    for (int nt = 0; nt < 4; ++nt) {
      int n = n0 + nt * 16 + ln;
      float bias = bo[n];
#pragma unroll
      for (int r = 0; r < 8; ++r) {
        int m = m0 + mi * 16 + r + 8 * half;
        out[(size_t)m * DDIM + n] = acc[mi][nt][r] + bias;
      }
    }
  }
}

// ---------------------------------------------------------------------------
extern "C" void kernel_launch(void* const* d_in, const int* in_sizes, int n_in,
                              void* d_out, int out_size, void* d_ws, size_t ws_size,
                              hipStream_t stream) {
  const float* x    = (const float*)d_in[0];
  const float* Wqkv = (const float*)d_in[1];
  const float* bqkv = (const float*)d_in[2];
  const float* Wo   = (const float*)d_in[3];
  const float* bo   = (const float*)d_in[4];
  float* out = (float*)d_out;

  char* ws = (char*)d_ws;
  const size_t nX  = (size_t)BDIM * SDIM * DDIM;  // 4,194,304
  const size_t nWq = (size_t)3 * DDIM * DDIM;     // 3,145,728
  const size_t nWo = (size_t)DDIM * DDIM;         // 1,048,576

  bf16* xb   = (bf16*)(ws);
  bf16* wqb  = (bf16*)(ws + 2 * nX);
  bf16* wob  = (bf16*)(ws + 2 * (nX + nWq));
  bf16* Qh   = (bf16*)(ws + 2 * (nX + nWq + nWo));
  bf16* Kh   = Qh + nX;
  bf16* Vth  = Kh + nX;
  bf16* vals = Vth + nX;

  cvt_f32_to_bf16<<<(unsigned)(nX / 1024), 256, 0, stream>>>(x, xb, (int)nX);
  cvt_f32_to_bf16<<<(unsigned)(nWq / 1024), 256, 0, stream>>>(Wqkv, wqb, (int)nWq);
  cvt_f32_to_bf16<<<(unsigned)(nWo / 1024), 256, 0, stream>>>(Wo, wob, (int)nWo);

  qkv_gemm<<<dim3(32, 24), 256, 0, stream>>>(xb, wqb, bqkv, Qh, Kh, Vth);
  attn_fwd<<<BDIM * HDIM * (SDIM / 64), 128, 0, stream>>>(Qh, Kh, Vth, vals);
  out_proj<<<dim3(32, 8), 256, 0, stream>>>(vals, wob, bo, out);
}
